// KCL_69784628626020
// MI455X (gfx1250) — matrix operand order repaired and checked by reading, not compiled, not run on hardware
//
#include <hip/hip_runtime.h>
#include <math.h>

#define N_    8192
#define D_    512
#define K_    8
#define NCLS  1000
#define TAU_INV 14.2857142857142857f   // 1/0.07
#define SPAD  (D_ + 8)                  // LDS row pad: +16B shifts banks per row

typedef __attribute__((ext_vector_type(16))) __bf16 v16bf;
typedef __attribute__((ext_vector_type(8)))  float  v8f;
typedef __attribute__((ext_vector_type(4)))  unsigned int v4u;

union BF16Frag { v16bf bf; v4u u[2]; };

// ---- zero class counts -----------------------------------------------------
__global__ void k_init(int* counts) {
  int t = blockIdx.x * 256 + threadIdx.x;
  if (t < NCLS) counts[t] = 0;
}

// ---- bincount(y) -----------------------------------------------------------
__global__ void k_bincount(const int* __restrict__ y, int* __restrict__ counts) {
  int i = blockIdx.x * 256 + threadIdx.x;
  if (i < N_) atomicAdd(&counts[y[i]], 1);
}

// ---- fp32 -> bf16 hi/lo split ---------------------------------------------
__global__ void k_convert(const float* __restrict__ q,
                          __bf16* __restrict__ qhi, __bf16* __restrict__ qlo) {
  int i = blockIdx.x * 256 + threadIdx.x;      // over N*D
  float x = q[i];
  __bf16 h = (__bf16)x;
  qhi[i] = h;
  qlo[i] = (__bf16)(x - (float)h);
}

// ---- colw[j] = 1 / counts[y[j]] -------------------------------------------
__global__ void k_colw(const int* __restrict__ y, const int* __restrict__ counts,
                       float* __restrict__ colw) {
  int i = blockIdx.x * 256 + threadIdx.x;
  if (i < N_) colw[i] = 1.0f / (float)counts[y[i]];
}

// ---- positives: psum[i] = sum_k exp(<q_i, k_ik>/tau) -----------------------
__global__ void k_pos(const float* __restrict__ q, const float* __restrict__ kmat,
                      float* __restrict__ psum) {
  __shared__ float sh[K_];
  int i    = blockIdx.x;
  int wave = threadIdx.x >> 5;
  int lane = threadIdx.x & 31;
  const float* qi = q + (size_t)i * D_;
  const float* kk = kmat + ((size_t)i * K_ + wave) * D_;
  float d = 0.f;
  #pragma unroll 4
  for (int t = lane; t < D_; t += 32) d += qi[t] * kk[t];
  #pragma unroll
  for (int off = 16; off; off >>= 1) d += __shfl_xor(d, off, 32);
  if (lane == 0) sh[wave] = __expf(d * TAU_INV);
  __syncthreads();
  if (threadIdx.x == 0) {
    float s = 0.f;
    #pragma unroll
    for (int w = 0; w < K_; ++w) s += sh[w];
    psum[i] = s;
  }
}

// ---- fused  S = (q q^T)/tau  +  masked num/den partial reductions ----------
// grid.x: 32-row strip (two A row-tiles in LDS); grid.y: column half.
// Each wave computes TWO 16x16 tiles sharing one B fragment -> halves L2 loads
// per WMMA. bf16x3: dot = hi*hi + hi*lo + lo*hi (fp32-class accuracy).
__global__ void __launch_bounds__(256) k_main(
    const __bf16* __restrict__ qhi, const __bf16* __restrict__ qlo,
    const int*    __restrict__ y,   const float* __restrict__ colw,
    float* __restrict__ partialN, float* __restrict__ partialD) {
  __shared__ __bf16 sAhi[32 * SPAD];
  __shared__ __bf16 sAlo[32 * SPAD];
  __shared__ float  sPartN[8][32];
  __shared__ float  sPartD[8][32];

  const int rb    = blockIdx.x * 32;
  const int cbase = blockIdx.y * (N_ / 32);   // 256 column tiles per half
  const int tid   = threadIdx.x;
  const int wave  = tid >> 5;
  const int lane  = tid & 31;
  const int hw    = (lane < 16) ? 0 : 1;      // half-wave
  const int nIdx  = lane & 15;

  // Stage the 32-row A strip (hi+lo) into padded LDS: 2048 x 16B chunks each.
  for (int c = tid; c < 2048; c += 256) {
    int row = c >> 6;           // 64 chunks per row (512 bf16 = 1024B)
    int off = c & 63;           // 16B units
    ((v4u*)(sAhi + row * SPAD))[off] =
        ((const v4u*)(qhi + (size_t)(rb + row) * D_))[off];
    ((v4u*)(sAlo + row * SPAD))[off] =
        ((const v4u*)(qlo + (size_t)(rb + row) * D_))[off];
  }
  __syncthreads();

  // Per-accumulator-register row ids / labels (ISA C-layout: reg r -> M=r|r+8)
  int rowi0[8], yi0[8], rowi1[8], yi1[8];
  #pragma unroll
  for (int r = 0; r < 8; ++r) {
    rowi0[r] = rb + r + hw * 8;       yi0[r] = y[rowi0[r]];
    rowi1[r] = rb + 16 + r + hw * 8;  yi1[r] = y[rowi1[r]];
  }

  float numAcc0[8] = {0,0,0,0,0,0,0,0}, denAcc0[8] = {0,0,0,0,0,0,0,0};
  float numAcc1[8] = {0,0,0,0,0,0,0,0}, denAcc1[8] = {0,0,0,0,0,0,0,0};

  const __bf16* ahiP0 = sAhi + (size_t)nIdx * SPAD + hw * 8;          // tile 0
  const __bf16* aloP0 = sAlo + (size_t)nIdx * SPAD + hw * 8;
  const __bf16* ahiP1 = sAhi + (size_t)(nIdx + 16) * SPAD + hw * 8;   // tile 1
  const __bf16* aloP1 = sAlo + (size_t)(nIdx + 16) * SPAD + hw * 8;

  for (int ct = cbase + wave; ct < cbase + N_ / 32; ct += 8) {
    const int col = ct * 16 + nIdx;                  // B: col = lane%16
    const __bf16* bhiP = qhi + (size_t)col * D_ + hw * 8;
    const __bf16* bloP = qlo + (size_t)col * D_ + hw * 8;

    // warm L2/L0 for the next column tile's B rows (global_prefetch_b8)
    if (ct + 8 < cbase + N_ / 32) {
      const size_t nb = (size_t)((ct + 8) * 16 + nIdx) * D_ + hw * 16;
      __builtin_prefetch(qhi + nb, 0, 1);
      __builtin_prefetch(qlo + nb, 0, 1);
    }

    v8f c0 = {0.f,0.f,0.f,0.f,0.f,0.f,0.f,0.f};
    v8f c1 = {0.f,0.f,0.f,0.f,0.f,0.f,0.f,0.f};
    #pragma unroll 2
    for (int kb = 0; kb < D_; kb += 32) {
      BF16Frag ah0, al0, ah1, al1, bh, bl;
      ah0.u[0] = *(const v4u*)(ahiP0 + kb);        // K = kb+{0..7 | 8..15}
      ah0.u[1] = *(const v4u*)(ahiP0 + kb + 16);   // K = kb+16+{0..7 | 8..15}
      al0.u[0] = *(const v4u*)(aloP0 + kb);
      al0.u[1] = *(const v4u*)(aloP0 + kb + 16);
      ah1.u[0] = *(const v4u*)(ahiP1 + kb);
      ah1.u[1] = *(const v4u*)(ahiP1 + kb + 16);
      al1.u[0] = *(const v4u*)(aloP1 + kb);
      al1.u[1] = *(const v4u*)(aloP1 + kb + 16);
      bh.u[0]  = *(const v4u*)(bhiP + kb);         // K = kb+{0..7 | 8..15}
      bh.u[1]  = *(const v4u*)(bhiP + kb + 16);
      bl.u[0]  = *(const v4u*)(bloP + kb);
      bl.u[1]  = *(const v4u*)(bloP + kb + 16);
      c0 = __builtin_amdgcn_wmma_f32_16x16x32_bf16(false, ah0.bf, false, bh.bf,
                                                   (short)0, c0, false, false);
      c0 = __builtin_amdgcn_wmma_f32_16x16x32_bf16(false, ah0.bf, false, bl.bf,
                                                   (short)0, c0, false, false);
      c0 = __builtin_amdgcn_wmma_f32_16x16x32_bf16(false, al0.bf, false, bh.bf,
                                                   (short)0, c0, false, false);
      c1 = __builtin_amdgcn_wmma_f32_16x16x32_bf16(false, ah1.bf, false, bh.bf,
                                                   (short)0, c1, false, false);
      c1 = __builtin_amdgcn_wmma_f32_16x16x32_bf16(false, ah1.bf, false, bl.bf,
                                                   (short)0, c1, false, false);
      c1 = __builtin_amdgcn_wmma_f32_16x16x32_bf16(false, al1.bf, false, bh.bf,
                                                   (short)0, c1, false, false);
    }

    const int   yj = y[col];
    const float wj = colw[col];
    #pragma unroll
    for (int r = 0; r < 8; ++r) {
      float s0 = c0[r] * TAU_INV;
      if (yi0[r] == yj) {       // in-batch positive (excl self), masked_fill(==0)
        if ((rowi0[r] != col) && (s0 != 0.f)) numAcc0[r] += __expf(s0);
      } else {
        if (s0 != 0.f) denAcc0[r] += __expf(s0) * wj;
      }
      float s1 = c1[r] * TAU_INV;
      if (yi1[r] == yj) {
        if ((rowi1[r] != col) && (s1 != 0.f)) numAcc1[r] += __expf(s1);
      } else {
        if (s1 != 0.f) denAcc1[r] += __expf(s1) * wj;
      }
    }
  }

  // reduce each accumulator across its 16-lane half (rows constant there)
  #pragma unroll
  for (int r = 0; r < 8; ++r) {
    float n0 = numAcc0[r], d0 = denAcc0[r], n1 = numAcc1[r], d1 = denAcc1[r];
    #pragma unroll
    for (int off = 8; off; off >>= 1) {
      n0 += __shfl_xor(n0, off, 32);  d0 += __shfl_xor(d0, off, 32);
      n1 += __shfl_xor(n1, off, 32);  d1 += __shfl_xor(d1, off, 32);
    }
    if (nIdx == 0) {
      sPartN[wave][r + hw * 8]      = n0;  sPartD[wave][r + hw * 8]      = d0;
      sPartN[wave][16 + r + hw * 8] = n1;  sPartD[wave][16 + r + hw * 8] = d1;
    }
  }
  __syncthreads();

  if (tid < 32) {                       // fixed-order combine -> deterministic
    int   row = rb + tid;
    float nT = 0.f, dT = 0.f;
    #pragma unroll
    for (int w = 0; w < 8; ++w) { nT += sPartN[w][tid]; dT += sPartD[w][tid]; }
    partialN[(size_t)blockIdx.y * N_ + row] = nT;
    partialD[(size_t)blockIdx.y * N_ + row] = dT;
  }
}

// ---- combine column-halves into per-row loss (deterministic) ---------------
__global__ void k_combine(const float* __restrict__ partialN,
                          const float* __restrict__ partialD,
                          const float* __restrict__ psum,
                          const int*   __restrict__ y,
                          const int*   __restrict__ counts,
                          float* __restrict__ lossRow) {
  int i = blockIdx.x * 256 + threadIdx.x;
  if (i >= N_) return;
  float numT = psum[i] + partialN[i] + partialN[N_ + i];
  float denT = partialD[i] + partialD[N_ + i];
  float cnt  = (float)(counts[y[i]] - 1 + K_);   // same-class (excl self) + K
  lossRow[i] = -(__logf(numT) - __logf(denT)) / cnt;
}

// ---- deterministic mean ----------------------------------------------------
__global__ void k_final(const float* __restrict__ lossRow, float* __restrict__ out) {
  __shared__ float red[256];
  int tid = threadIdx.x;
  float s = 0.f;
  for (int i = tid; i < N_; i += 256) s += lossRow[i];
  red[tid] = s;
  __syncthreads();
  for (int off = 128; off; off >>= 1) {
    if (tid < off) red[tid] += red[tid + off];
    __syncthreads();
  }
  if (tid == 0) out[0] = red[0] / (float)N_;
}

extern "C" void kernel_launch(void* const* d_in, const int* in_sizes, int n_in,
                              void* d_out, int out_size, void* d_ws, size_t ws_size,
                              hipStream_t stream) {
  (void)in_sizes; (void)n_in; (void)out_size; (void)ws_size;
  const float* q    = (const float*)d_in[0];   // [N, D]
  const float* kmat = (const float*)d_in[1];   // [N, K, D]
  const int*   y    = (const int*)d_in[2];     // [N]
  float* out = (float*)d_out;

  char* ws = (char*)d_ws;
  int*    counts   = (int*)(ws + 0);                          // 4000 B
  float*  colw     = (float*)(ws + 4096);                     // 32 KB
  float*  psum     = (float*)(ws + 36864);                    // 32 KB
  float*  lossRow  = (float*)(ws + 69632);                    // 32 KB
  float*  partialN = (float*)(ws + 102400);                   // 64 KB (2 x N)
  float*  partialD = (float*)(ws + 167936);                   // 64 KB (2 x N)
  __bf16* qhi      = (__bf16*)(ws + 232448);                  // 8 MB
  __bf16* qlo      = (__bf16*)(ws + 232448 + (size_t)N_ * D_ * 2); // 8 MB

  dim3 mainGrid(N_ / 32, 2);
  k_init    <<<4,             256, 0, stream>>>(counts);
  k_bincount<<<N_ / 256,      256, 0, stream>>>(y, counts);
  k_convert <<<(N_ * D_)/256, 256, 0, stream>>>(q, qhi, qlo);
  k_colw    <<<N_ / 256,      256, 0, stream>>>(y, counts, colw);
  k_pos     <<<N_,            256, 0, stream>>>(q, kmat, psum);
  k_main    <<<mainGrid,      256, 0, stream>>>(qhi, qlo, y, colw, partialN, partialD);
  k_combine <<<N_ / 256,      256, 0, stream>>>(partialN, partialD, psum, y, counts, lossRow);
  k_final   <<<1,             256, 0, stream>>>(lossRow, out);
}
